// GATFeat_66675072303440
// MI455X (gfx1250) — matrix operand order repaired and checked by reading, not compiled, not run on hardware
//
#include <hip/hip_runtime.h>
#include <hip/hip_bf16.h>

// ---------------------------------------------------------------------------
// Problem constants (match reference)
// ---------------------------------------------------------------------------
#define BB     4
#define NN_    2048
#define NNODE  (BB * NN_)        // 8192
#define DEG    16
#define EEDGE  (NNODE * DEG)     // 131072 random edges
#define ETOT   (EEDGE + NNODE)   // + self loops = 139264
#define OUTC   1792              // 256 + 512 + 1024
#define NEG_SLOPE 0.2f

typedef __attribute__((ext_vector_type(16))) __bf16 v16bf;
typedef __attribute__((ext_vector_type(8)))  float  v8f;

// round-to-nearest-even f32 -> bf16
__device__ __forceinline__ unsigned short f32_to_bf16(float f) {
    unsigned int u = __float_as_uint(f);
    unsigned int r = u + 0x7FFFu + ((u >> 16) & 1u);
    return (unsigned short)(r >> 16);
}

__device__ __forceinline__ void atomicMaxF(float* addr, float val) {
    int* ia = (int*)addr;
    int cur = *ia;
    while (__int_as_float(cur) < val) {
        int prev = atomicCAS(ia, cur, __float_as_int(val));
        if (prev == cur) break;
        cur = prev;
    }
}

// A-fragment packed index for activation element (node, c) with fin = ktiles*32:
// layout [mt][kt][lane 0..31][16 bf16] so one wave K-step is one contiguous
// 32-byte object per lane.  ISA A-layout: lane hi-half selects K sub-range.
__device__ __forceinline__ size_t a_pack_index(int node, int c, int ktiles) {
    int mt  = node >> 4;
    int lm  = node & 15;
    int kt  = c >> 5;
    int rem = c & 31;
    int h   = (rem >> 3) & 1;                    // lane high-half
    int e   = (rem & 7) + ((rem & 16) ? 8 : 0);  // element within fragment
    int lane = h * 16 + lm;
    return (((size_t)mt * ktiles + kt) * 32 + lane) * 16 + e;
}

// ---------------------------------------------------------------------------
// Fill kernel (init accumulators / segment-max / segment-sum)
// ---------------------------------------------------------------------------
__global__ void fill_f32(float* __restrict__ p, float v, int count) {
    int t = blockIdx.x * blockDim.x + threadIdx.x;
    if (t < count) p[t] = v;
}

// ---------------------------------------------------------------------------
// Input fusion: per-node Conv1d(k=1) 3->64 and 32->64, relu, concat -> 128.
// Writes f32 feature and the A-fragment-packed bf16 copy (fin=128, ktiles=4).
// ---------------------------------------------------------------------------
__global__ void fuse_kernel(const float* __restrict__ x,
                            const float* __restrict__ emb,
                            const float* __restrict__ g_w,  // [3,64]
                            const float* __restrict__ g_b,
                            const float* __restrict__ c_w,  // [32,64]
                            const float* __restrict__ c_b,
                            float* __restrict__ featf,        // [NNODE,128]
                            unsigned short* __restrict__ featbf) {
    int t = blockIdx.x * blockDim.x + threadIdx.x;
    if (t >= NNODE * 128) return;
    int node = t >> 7;
    int o    = t & 127;
    int b = node / NN_;
    int i = node - b * NN_;
    float acc;
    if (o < 64) {
        acc = g_b[o];
        #pragma unroll
        for (int c = 0; c < 3; ++c)
            acc += x[(size_t)(b * 3 + c) * NN_ + i] * g_w[c * 64 + o];
    } else {
        int o2 = o - 64;
        acc = c_b[o2];
        #pragma unroll
        for (int c = 0; c < 32; ++c)
            acc += emb[(size_t)(b * 32 + c) * NN_ + i] * c_w[c * 64 + o2];
    }
    acc = fmaxf(acc, 0.0f);
    featf[t] = acc;
    featbf[a_pack_index(node, o, 128 >> 5)] = f32_to_bf16(acc);
}

// ---------------------------------------------------------------------------
// Pack [Wl | Wr] (f32, [fin,fout] each, row-major) into WMMA B-fragment
// order, bf16, grouped so one wave's K-step is fully contiguous:
//   layout [ng][kt][j 0..3][lane 0..31][16 bf16]       (ng = nt/4, j = nt%4)
// For lane L of tile nt: column N = nt*16 + (L&15),
// K-base = kt*32 + (L>>4)*8, elements e=0..15 are W[kbase+e+(e>=8?8:0)][N].
// ---------------------------------------------------------------------------
__global__ void pack_w_kernel(const float* __restrict__ Wl,
                              const float* __restrict__ Wr,
                              unsigned short* __restrict__ out,
                              int fin, int fout) {
    int ktiles = fin >> 5;
    int fout2  = fout * 2;
    int ntiles = fout2 >> 4;
    int total  = ntiles * ktiles * 32;
    int t = blockIdx.x * blockDim.x + threadIdx.x;
    if (t >= total) return;
    int lane = t & 31;
    int kt   = (t >> 5) % ktiles;
    int nt   = (t >> 5) / ktiles;
    int col  = nt * 16 + (lane & 15);
    int kb   = kt * 32 + (lane >> 4) * 8;
    const float* W;
    int c;
    if (col < fout) { W = Wl; c = col; } else { W = Wr; c = col - fout; }
    int ng = nt >> 2, j = nt & 3;
    size_t slot = ((((size_t)ng * ktiles + kt) * 4 + j) * 32 + lane) * 16;
    unsigned short* dst = out + slot;
    #pragma unroll
    for (int e = 0; e < 16; ++e)
        dst[e] = f32_to_bf16(W[(size_t)(kb + e + (e >= 8 ? 8 : 0)) * fout + c]);
}

// ---------------------------------------------------------------------------
// WMMA GEMM: C[n, 2*fout] = A(packed bf16) @ [Wl|Wr](packed bf16) + bias.
// One wave -> 16 rows x 64 cols (4 accumulators).  Every fragment is one
// contiguous v16bf load (-> two adjacent b128 loads, no shuffles/copies).
// K loop unrolled by 2 with ping-pong register sets.  Uniform control flow
// only: EXEC stays all-ones as WMMA requires.
// ---------------------------------------------------------------------------
__global__ __launch_bounds__(256)
void gemm_wmma_kernel(const __bf16* __restrict__ Apk,   // packed A fragments
                      const __bf16* __restrict__ Wpk,   // packed B fragments
                      const float*  __restrict__ bl,    // [fout]
                      const float*  __restrict__ br,    // [fout]
                      float* __restrict__ C,            // [nrows, 2*fout]
                      int nrows, int fin, int fout) {
    const int fout2  = fout * 2;
    const int ktiles = fin >> 5;
    const int mtiles = nrows >> 4;
    const int lane   = threadIdx.x & 31;
    const int wave   = blockIdx.x * (blockDim.x >> 5) + (threadIdx.x >> 5);
    const int mt     = wave % mtiles;
    const int ng     = wave / mtiles;       // group of 4 n-tiles
    const int hi     = lane >> 4;
    const int lm     = lane & 15;

    // A: one K-step chunk = 32 lanes * 16 = 512 elems (1024 B)
    const __bf16* Ap = Apk + ((size_t)mt * ktiles * 32 + lane) * 16;
    // B: one K-step chunk = 4*32*16 = 2048 elems (4096 B)
    const __bf16* Bp = Wpk + (size_t)ng * ktiles * 2048 + lane * 16;

    v8f acc[4];
    #pragma unroll
    for (int j = 0; j < 4; ++j) acc[j] = (v8f){};

    // Ping-pong fragment sets (each one contiguous v16bf object)
    v16bf aF[2], bF[2][4];

    // Prologue: load set 0 (K step 0)
    aF[0] = *(const v16bf*)(Ap);
    #pragma unroll
    for (int j = 0; j < 4; ++j)
        bF[0][j] = *(const v16bf*)(Bp + j * 512);

    for (int kt = 0; kt < ktiles; kt += 2) {
        // Prefetch K step kt+1 into set 1 (kt+1 <= ktiles-1: always valid)
        aF[1] = *(const v16bf*)(Ap + 512);
        #pragma unroll
        for (int j = 0; j < 4; ++j)
            bF[1][j] = *(const v16bf*)(Bp + 2048 + j * 512);

        // Compute K step kt (set 0)
        #pragma unroll
        for (int j = 0; j < 4; ++j)
            acc[j] = __builtin_amdgcn_wmma_f32_16x16x32_bf16(
                false, aF[0], false, bF[0][j], (short)0, acc[j], false, false);

        // Prefetch K step kt+2 into set 0 (clamped at the tail; uniform)
        const __bf16* Ap2 = (kt + 2 < ktiles) ? (Ap + 1024) : Ap;
        const __bf16* Bp2 = (kt + 2 < ktiles) ? (Bp + 4096) : Bp;
        aF[0] = *(const v16bf*)(Ap2);
        #pragma unroll
        for (int j = 0; j < 4; ++j)
            bF[0][j] = *(const v16bf*)(Bp2 + j * 512);

        // Compute K step kt+1 (set 1)
        #pragma unroll
        for (int j = 0; j < 4; ++j)
            acc[j] = __builtin_amdgcn_wmma_f32_16x16x32_bf16(
                false, aF[1], false, bF[1][j], (short)0, acc[j], false, false);

        Ap = Ap2; Bp = Bp2;
    }

    // Epilogue: D VGPR r holds row M = r + 8*hi, column N = lane&15.
    #pragma unroll
    for (int j = 0; j < 4; ++j) {
        int col = (ng * 4 + j) * 16 + lm;
        float bias = (col < fout) ? bl[col] : br[col - fout];
        #pragma unroll
        for (int r = 0; r < 8; ++r) {
            int row = mt * 16 + hi * 8 + r;
            C[(size_t)row * fout2 + col] = acc[j][r] + bias;
        }
    }
}

// ---------------------------------------------------------------------------
// Edge pass 1: logit e = att . leakyrelu(xl[src] + xr[dst]); segment max.
// One wave per edge.
// ---------------------------------------------------------------------------
__global__ __launch_bounds__(256)
void edge_logit_kernel(const long long* __restrict__ ei,
                       const float* __restrict__ xlr,  // [NNODE, 2*fout]
                       const float* __restrict__ att,  // [fout]
                       float* __restrict__ eout,       // [ETOT]
                       float* __restrict__ segmax,     // [NNODE]
                       int fout) {
    int lane = threadIdx.x & 31;
    int edge = blockIdx.x * (blockDim.x >> 5) + (threadIdx.x >> 5);
    if (edge >= ETOT) return;
    int s, d;
    if (edge < EEDGE) { s = (int)ei[edge]; d = (int)ei[EEDGE + edge]; }
    else              { s = d = edge - EEDGE; }
    const int fout2 = fout * 2;
    const float* xs = xlr + (size_t)s * fout2;
    const float* xd = xlr + (size_t)d * fout2 + fout;
    float p = 0.0f;
    for (int c = lane; c < fout; c += 32) {
        float v = xs[c] + xd[c];
        v = (v > 0.0f) ? v : NEG_SLOPE * v;
        p += v * att[c];
    }
    #pragma unroll
    for (int o = 16; o > 0; o >>= 1) p += __shfl_xor(p, o, 32);
    if (lane == 0) {
        eout[edge] = p;
        atomicMaxF(&segmax[d], p);
    }
}

// ---------------------------------------------------------------------------
// Edge pass 2: a = exp(e - max[dst]); segment sum.  One thread per edge.
// ---------------------------------------------------------------------------
__global__ void edge_exp_kernel(const long long* __restrict__ ei,
                                const float* __restrict__ eout,
                                const float* __restrict__ segmax,
                                float* __restrict__ aout,
                                float* __restrict__ segsum) {
    int edge = blockIdx.x * blockDim.x + threadIdx.x;
    if (edge >= ETOT) return;
    int d = (edge < EEDGE) ? (int)ei[EEDGE + edge] : (edge - EEDGE);
    float a = __expf(eout[edge] - segmax[d]);
    aout[edge] = a;
    atomicAdd(&segsum[d], a);
}

// ---------------------------------------------------------------------------
// Edge pass 3: out[dst] += (a/segsum[dst]) * xl[src].  One wave per edge.
// ---------------------------------------------------------------------------
__global__ __launch_bounds__(256)
void edge_aggr_kernel(const long long* __restrict__ ei,
                      const float* __restrict__ xlr,
                      const float* __restrict__ aout,
                      const float* __restrict__ segsum,
                      float* __restrict__ agg,          // [NNODE, fout]
                      int fout) {
    int lane = threadIdx.x & 31;
    int edge = blockIdx.x * (blockDim.x >> 5) + (threadIdx.x >> 5);
    if (edge >= ETOT) return;
    int s, d;
    if (edge < EEDGE) { s = (int)ei[edge]; d = (int)ei[EEDGE + edge]; }
    else              { s = d = edge - EEDGE; }
    float alpha = aout[edge] / segsum[d];
    const int fout2 = fout * 2;
    const float* xs = xlr + (size_t)s * fout2;
    float* od = agg + (size_t)d * fout;
    for (int c = lane; c < fout; c += 32)
        atomicAdd(&od[c], alpha * xs[c]);
}

// ---------------------------------------------------------------------------
// Node post-pass: + bias, optional relu; write f32 feat (in-place), packed
// bf16 feat for the next GEMM (ktiles of the NEXT layer = fout/32), and the
// d_out slice for layers 1/2/4.
// ---------------------------------------------------------------------------
__global__ void node_post_kernel(float* __restrict__ featf,          // [NNODE,fout]
                                 unsigned short* __restrict__ featbf,
                                 const float* __restrict__ bias,
                                 float* __restrict__ d_out,
                                 int fout, int relu, int out_off) {
    int t = blockIdx.x * blockDim.x + threadIdx.x;
    if (t >= NNODE * fout) return;
    int node = t / fout;
    int c    = t - node * fout;
    float v = featf[t] + bias[c];
    if (relu) v = fmaxf(v, 0.0f);
    featf[t] = v;
    featbf[a_pack_index(node, c, fout >> 5)] = f32_to_bf16(v);
    if (out_off >= 0)
        d_out[(size_t)node * OUTC + out_off + c] = v;
}

// ---------------------------------------------------------------------------
// Host launcher
// ---------------------------------------------------------------------------
extern "C" void kernel_launch(void* const* d_in, const int* in_sizes, int n_in,
                              void* d_out, int out_size, void* d_ws, size_t ws_size,
                              hipStream_t stream) {
    const float*     x   = (const float*)d_in[0];
    const float*     emb = (const float*)d_in[1];
    const long long* ei  = (const long long*)d_in[2];
    const float*     g_w = (const float*)d_in[3];
    const float*     g_b = (const float*)d_in[4];
    const float*     c_w = (const float*)d_in[5];
    const float*     c_b = (const float*)d_in[6];

    const int fins[4]  = {128, 256, 512, 1024};
    const int fouts[4] = {256, 512, 1024, 1024};
    const int relus[4]   = {1, 1, 1, 0};
    const int outoffs[4] = {0, 256, -1, 768};

    const float* Wl[4]; const float* bl[4];
    const float* Wr[4]; const float* br[4];
    const float* att[4]; const float* lbias[4];
    for (int l = 0; l < 4; ++l) {
        int b = 7 + l * 6;
        Wl[l]    = (const float*)d_in[b + 0];
        bl[l]    = (const float*)d_in[b + 1];
        Wr[l]    = (const float*)d_in[b + 2];
        br[l]    = (const float*)d_in[b + 3];
        att[l]   = (const float*)d_in[b + 4];
        lbias[l] = (const float*)d_in[b + 5];
    }

    // ---- workspace carving ----
    char* ws = (char*)d_ws;
    size_t off = 0;
    auto carve = [&](size_t bytes) {
        size_t o = off;
        off = (off + bytes + 255) & ~(size_t)255;
        return o;
    };
    size_t pkOff[4];
    for (int l = 0; l < 4; ++l)
        pkOff[l] = carve((size_t)fins[l] * fouts[l] * 2 /*Wl|Wr*/ * 2 /*bf16*/);
    size_t featfOff  = carve((size_t)NNODE * 1024 * 4);
    size_t featbfOff = carve((size_t)NNODE * 1024 * 2);
    size_t xlrOff    = carve((size_t)NNODE * 2048 * 4);
    size_t eOff      = carve((size_t)ETOT * 4);
    size_t aOff      = carve((size_t)ETOT * 4);
    size_t mOff      = carve((size_t)NNODE * 4);
    size_t sOff      = carve((size_t)NNODE * 4);

    float*          featf  = (float*)(ws + featfOff);
    unsigned short* featbf = (unsigned short*)(ws + featbfOff);
    float*          xlr    = (float*)(ws + xlrOff);
    float*          ebuf   = (float*)(ws + eOff);
    float*          abuf   = (float*)(ws + aOff);
    float*          segmax = (float*)(ws + mOff);
    float*          segsum = (float*)(ws + sOff);
    float*          out    = (float*)d_out;

    // ---- weight packing (deterministic, every call) ----
    for (int l = 0; l < 4; ++l) {
        int total = (fouts[l] * 2 / 16) * (fins[l] / 32) * 32;
        pack_w_kernel<<<(total + 255) / 256, 256, 0, stream>>>(
            Wl[l], Wr[l], (unsigned short*)(ws + pkOff[l]), fins[l], fouts[l]);
    }

    // ---- input fusion ----
    {
        int total = NNODE * 128;
        fuse_kernel<<<(total + 255) / 256, 256, 0, stream>>>(
            x, emb, g_w, g_b, c_w, c_b, featf, featbf);
    }

    // ---- layers ----
    for (int l = 0; l < 4; ++l) {
        const int fin = fins[l], fout = fouts[l], fout2 = fout * 2;

        // GEMM: xlr = feat @ [Wl|Wr] + [bl|br]
        {
            int waves  = (NNODE / 16) * (fout2 / 64);
            int blocks = waves / 8;
            gemm_wmma_kernel<<<blocks, 256, 0, stream>>>(
                (const __bf16*)featbf, (const __bf16*)(ws + pkOff[l]),
                bl[l], br[l], xlr, NNODE, fin, fout);
        }

        // init segment max / sum / aggregation accumulator (feat is dead now)
        fill_f32<<<(NNODE + 255) / 256, 256, 0, stream>>>(segmax, -1e30f, NNODE);
        fill_f32<<<(NNODE + 255) / 256, 256, 0, stream>>>(segsum, 0.0f, NNODE);
        {
            int total = NNODE * fout;
            fill_f32<<<(total + 255) / 256, 256, 0, stream>>>(featf, 0.0f, total);
        }

        // edge phases
        {
            int blocks = (ETOT + 7) / 8;
            edge_logit_kernel<<<blocks, 256, 0, stream>>>(
                ei, xlr, att[l], ebuf, segmax, fout);
        }
        edge_exp_kernel<<<(ETOT + 255) / 256, 256, 0, stream>>>(
            ei, ebuf, segmax, abuf, segsum);
        {
            int blocks = (ETOT + 7) / 8;
            edge_aggr_kernel<<<blocks, 256, 0, stream>>>(
                ei, xlr, abuf, segsum, featf, fout);
        }

        // node post: bias (+relu), write next feat + output slice
        {
            int total = NNODE * fout;
            node_post_kernel<<<(total + 255) / 256, 256, 0, stream>>>(
                featf, featbf, lbias[l], out, fout, relus[l], outoffs[l]);
        }
    }
}
